// GruBlock_19353122636519
// MI455X (gfx1250) — matrix operand order repaired and checked by reading, not compiled, not run on hardware
//
#include <hip/hip_runtime.h>
#include <math.h>

typedef __attribute__((ext_vector_type(2)))  float    v2f;
typedef __attribute__((ext_vector_type(4)))  float    f4;
typedef __attribute__((ext_vector_type(8)))  float    v8f;
typedef __attribute__((ext_vector_type(16))) __bf16   v16bf;
typedef __attribute__((ext_vector_type(2)))  __bf16   bf2;
typedef __attribute__((ext_vector_type(8)))  unsigned u8v;
typedef __attribute__((ext_vector_type(4)))  unsigned u4v;
typedef int i4g __attribute__((vector_size(4 * sizeof(int))));

#define B_ 128
#define T_ 512
#define K_ 1629
#define H_ 128
#define M_ (B_ * T_)          // 65536
#define KP_ 816               // ceil(K/2) k-pairs (zero padded)

#if __has_builtin(__builtin_amdgcn_global_load_async_to_lds_b128)
#define HAS_ASYNC_LDS 1
#else
#define HAS_ASYNC_LDS 0
#endif

// Async global->LDS 16B copy (falls back to a plain copy if builtin absent).
__device__ __forceinline__ void async_copy16(unsigned* ldsDst, const unsigned* gSrc)
{
#if HAS_ASYNC_LDS
    __builtin_amdgcn_global_load_async_to_lds_b128(
        (__attribute__((address_space(1))) i4g*)gSrc,
        (__attribute__((address_space(3))) i4g*)ldsDst, 0, 0);
#else
    *(u4v*)ldsDst = *(const u4v*)gSrc;
#endif
}

__device__ __forceinline__ void wait_async_all()
{
#if HAS_ASYNC_LDS
    asm volatile("s_wait_asynccnt 0x0" ::: "memory");
#endif
}

// ---------------------------------------------------------------------------
// Prep: split Wx (rows H..H+K-1 of each W) into bf16 hi/lo, packed as k-pairs
// per dword ([15:0]=even k, [31:16]=odd k), layout [gate][kpair][H].
// ---------------------------------------------------------------------------
__global__ __launch_bounds__(256)
void gru_prep_w(const float* __restrict__ Wr, const float* __restrict__ Wu,
                const float* __restrict__ Wc,
                unsigned* __restrict__ WhP, unsigned* __restrict__ WlP)
{
    int idx = blockIdx.x * 256 + threadIdx.x;
    const int total = 3 * KP_ * H_;
    if (idx >= total) return;
    int g   = idx / (KP_ * H_);
    int rem = idx - g * (KP_ * H_);
    int p   = rem >> 7;
    int h   = rem & 127;
    const float* W = (g == 0) ? Wr : (g == 1) ? Wu : Wc;
    int k0 = 2 * p, k1 = 2 * p + 1;
    float f0 = (k0 < K_) ? W[(size_t)(H_ + k0) * H_ + h] : 0.0f;
    float f1 = (k1 < K_) ? W[(size_t)(H_ + k1) * H_ + h] : 0.0f;
    __bf16 h0 = (__bf16)f0, h1 = (__bf16)f1;
    __bf16 l0 = (__bf16)(f0 - (float)h0), l1 = (__bf16)(f1 - (float)h1);
    bf2 hp = {h0, h1}, lp = {l0, l1};
    WhP[idx] = __builtin_bit_cast(unsigned, hp);
    WlP[idx] = __builtin_bit_cast(unsigned, lp);
}

// ---------------------------------------------------------------------------
// Phase 1: xg = x @ Wx_g + b_g for g in {r,u,c}; M=65536, N=384, K=1629.
// Split-bf16 (acc += Ah*Bh + Ah*Bl + Al*Bh) on v_wmma_f32_16x16x32_bf16.
// Block = 32 rows x 384 cols, 8 waves; wave = 2 M-tiles x 3 N-tiles.
// Double-buffered LDS; W tiles streamed via async global->LDS.
// ---------------------------------------------------------------------------
#define P1_MT 32
#define P1_KC 32              // k per chunk = 16 k-pairs
#define P1_XS 20              // 16 kpair-dwords + 4 pad (b128-aligned)
#define P1_WS 396             // 384 + 12 pad: 4*396 % 64 == 48 -> halves disjoint
#define P1_NCH ((K_ + P1_KC - 1) / P1_KC)   // 51

__device__ __forceinline__ v16bf loadA(const unsigned (*Xs)[P1_XS], int m, int lhi)
{
    u4v t0 = *(const u4v*)&Xs[m][lhi * 4];
    u4v t1 = *(const u4v*)&Xs[m][lhi * 4 + 8];
    u8v u;
    u[0] = t0[0]; u[1] = t0[1]; u[2] = t0[2]; u[3] = t0[3];
    u[4] = t1[0]; u[5] = t1[1]; u[6] = t1[2]; u[7] = t1[3];
    return __builtin_bit_cast(v16bf, u);
}

__device__ __forceinline__ v16bf loadB(const unsigned (*Ws)[P1_WS], int lhi, int colb)
{
    u8v u;
#pragma unroll
    for (int v = 0; v < 8; ++v) u[v] = Ws[(v < 4) ? (lhi * 4 + v) : (4 + lhi * 4 + v)][colb];
    return __builtin_bit_cast(v16bf, u);
}

__global__ __launch_bounds__(256)
void gru_phase1(const float* __restrict__ x,
                const float* __restrict__ br, const float* __restrict__ bu,
                const float* __restrict__ bc,
                const unsigned* __restrict__ WhP, const unsigned* __restrict__ WlP,
                float* __restrict__ xr, float* __restrict__ xu,
                float* __restrict__ xc)
{
    __shared__ unsigned XhS[2][P1_MT][P1_XS];
    __shared__ unsigned XlS[2][P1_MT][P1_XS];
    __shared__ unsigned WhS[2][16][P1_WS];
    __shared__ unsigned WlS[2][16][P1_WS];

    const int tid  = threadIdx.x;
    const int lane = tid & 31;
    const int wave = tid >> 5;         // 0..7 -> cols [wave*48, wave*48+48)
    const int m0   = blockIdx.x * P1_MT;
    const int l16  = lane & 15;
    const int lhi  = lane >> 4;

    // Per-thread X staging coordinates (2 slots of one k-pair each).
    const int xrow0 = tid >> 4;            // slot 0: rows 0..15
    const int xp0   = tid & 15;
    const int xrow1 = (tid + 256) >> 4;    // slot 1: rows 16..31
    const int xp1   = xp0;

    // Accumulators seeded with bias.
    v8f acc[2][3];
#pragma unroll
    for (int j = 0; j < 3; ++j) {
        int ncol = wave * 48 + j * 16 + l16;
        int g = ncol >> 7;
        int h = ncol & 127;
        float bias = (g == 0) ? br[h] : (g == 1) ? bu[h] : bc[h];
#pragma unroll
        for (int mt = 0; mt < 2; ++mt)
#pragma unroll
            for (int r = 0; r < 8; ++r) acc[mt][j][r] = bias;
    }

    // issue async copies of W chunk `ch` into buffer `buf`
    auto stageW = [&](int buf, int ch) {
        const int kb2 = ch * (P1_KC / 2);
#pragma unroll
        for (int g = 0; g < 3; ++g) {
            const unsigned* srcH = WhP + ((size_t)g * KP_ + kb2) * H_;
            const unsigned* srcL = WlP + ((size_t)g * KP_ + kb2) * H_;
#pragma unroll
            for (int v = 0; v < 2; ++v) {
                int idx = tid + v * 256;        // 0..511
                int row = idx >> 5;             // kpair row 0..15
                int c   = (idx & 31) * 4;       // dword col
                async_copy16(&WhS[buf][row][g * 128 + c], srcH + row * 128 + c);
                async_copy16(&WlS[buf][row][g * 128 + c], srcL + row * 128 + c);
            }
        }
    };
    // load X chunk `ch` into registers (4 floats / thread)
    float xf[4];
    auto loadX = [&](int ch) {
        const int kb = ch * P1_KC;
        const float* r0 = x + (size_t)(m0 + xrow0) * K_;
        const float* r1 = x + (size_t)(m0 + xrow1) * K_;
        int k0 = kb + 2 * xp0;
        xf[0] = (k0     < K_) ? r0[k0]     : 0.0f;
        xf[1] = (k0 + 1 < K_) ? r0[k0 + 1] : 0.0f;
        xf[2] = (k0     < K_) ? r1[k0]     : 0.0f;
        xf[3] = (k0 + 1 < K_) ? r1[k0 + 1] : 0.0f;
    };
    // split-convert X registers and store to buffer `buf`
    auto storeX = [&](int buf) {
        __bf16 h0 = (__bf16)xf[0], h1 = (__bf16)xf[1];
        __bf16 l0 = (__bf16)(xf[0] - (float)h0), l1 = (__bf16)(xf[1] - (float)h1);
        bf2 hp = {h0, h1}, lp = {l0, l1};
        XhS[buf][xrow0][xp0] = __builtin_bit_cast(unsigned, hp);
        XlS[buf][xrow0][xp0] = __builtin_bit_cast(unsigned, lp);
        __bf16 h2 = (__bf16)xf[2], h3 = (__bf16)xf[3];
        __bf16 l2 = (__bf16)(xf[2] - (float)h2), l3 = (__bf16)(xf[3] - (float)h3);
        bf2 hp2 = {h2, h3}, lp2 = {l2, l3};
        XhS[buf][xrow1][xp1] = __builtin_bit_cast(unsigned, hp2);
        XlS[buf][xrow1][xp1] = __builtin_bit_cast(unsigned, lp2);
    };

    // --- prologue: fill buffer 0 ---
    stageW(0, 0);
    loadX(0);
    storeX(0);
    wait_async_all();
    __syncthreads();

    for (int ch = 0; ch < P1_NCH; ++ch) {
        const int buf = ch & 1;

        // Start next chunk: async W -> LDS, X -> registers (overlaps WMMAs).
        if (ch + 1 < P1_NCH) {
            stageW(buf ^ 1, ch + 1);
            loadX(ch + 1);
            if (tid < 32)
                __builtin_prefetch(&x[(size_t)(m0 + tid) * K_ + (ch + 2) * P1_KC], 0, 1);
        }

        // --- fragments + 3-term bf16 WMMA on current buffer ---
        v16bf Ah0 = loadA(XhS[buf], 0 * 16 + l16, lhi);
        v16bf Al0 = loadA(XlS[buf], 0 * 16 + l16, lhi);
        v16bf Ah1 = loadA(XhS[buf], 1 * 16 + l16, lhi);
        v16bf Al1 = loadA(XlS[buf], 1 * 16 + l16, lhi);
#pragma unroll
        for (int j = 0; j < 3; ++j) {
            int colb = wave * 48 + j * 16 + l16;
            v16bf Bh = loadB(WhS[buf], lhi, colb);
            v16bf Bl = loadB(WlS[buf], lhi, colb);
            acc[0][j] = __builtin_amdgcn_wmma_f32_16x16x32_bf16(
                false, Al0, false, Bh, (short)0, acc[0][j], false, false);
            acc[0][j] = __builtin_amdgcn_wmma_f32_16x16x32_bf16(
                false, Ah0, false, Bl, (short)0, acc[0][j], false, false);
            acc[0][j] = __builtin_amdgcn_wmma_f32_16x16x32_bf16(
                false, Ah0, false, Bh, (short)0, acc[0][j], false, false);
            acc[1][j] = __builtin_amdgcn_wmma_f32_16x16x32_bf16(
                false, Al1, false, Bh, (short)0, acc[1][j], false, false);
            acc[1][j] = __builtin_amdgcn_wmma_f32_16x16x32_bf16(
                false, Ah1, false, Bl, (short)0, acc[1][j], false, false);
            acc[1][j] = __builtin_amdgcn_wmma_f32_16x16x32_bf16(
                false, Ah1, false, Bh, (short)0, acc[1][j], false, false);
        }

        // Finish next chunk's X staging, then make both buffers coherent.
        if (ch + 1 < P1_NCH) storeX(buf ^ 1);
        wait_async_all();      // async W writes to LDS complete
        __syncthreads();       // everyone done reading buf / writing buf^1
    }

    // --- store: gate from column; xr -> d_out (temp), xu/xc -> workspace ---
#pragma unroll
    for (int j = 0; j < 3; ++j) {
        int ncol = wave * 48 + j * 16 + l16;
        int g = ncol >> 7;
        int h = ncol & 127;
        float* outp = (g == 0) ? xr : (g == 1) ? xu : xc;
#pragma unroll
        for (int mt = 0; mt < 2; ++mt)
#pragma unroll
            for (int r = 0; r < 8; ++r) {
                int m = m0 + mt * 16 + r + lhi * 8;
                outp[(size_t)m * H_ + h] = acc[mt][j][r];
            }
    }
}

// ---------------------------------------------------------------------------
// Phase 2: sequential GRU recurrence. One persistent 1024-thread workgroup.
// Wh_r/u/c + h resident in LDS (~270KB of the 320KB WGP LDS). fp32 WMMA.
// ---------------------------------------------------------------------------
#define P2_WS 136   // 128 + 8 pad (paired-row B reads conflict-free)
#define P2_HS 133   // 128 + 5 pad (column A reads conflict-free)

__global__ __launch_bounds__(1024)
void gru_phase2(const float* __restrict__ Wr, const float* __restrict__ Wu,
                const float* __restrict__ Wc,
                const float* __restrict__ xu, const float* __restrict__ xc,
                float* __restrict__ out /* in: xr, out: h states */)
{
    __shared__ float Wrs[H_][P2_WS];
    __shared__ float Wus[H_][P2_WS];
    __shared__ float Wcs[H_][P2_WS];
    __shared__ float hs[B_][P2_HS];

    const int tid  = threadIdx.x;
    const int lane = tid & 31;
    const int wave = tid >> 5;
    const int l16  = lane & 15;
    const int lhi  = lane >> 4;
    const int mt   = wave >> 2;  // rows [mt*16, +16)
    const int nq   = wave & 3;   // cols [nq*32, +32)

#pragma unroll
    for (int v = 0; v < 4; ++v) {
        int idx = tid + v * 1024;
        int row = idx >> 5;
        int h0  = (idx & 31) * 4;
        *(f4*)&Wrs[row][h0] = *(const f4*)(Wr + (size_t)row * H_ + h0);
        *(f4*)&Wus[row][h0] = *(const f4*)(Wu + (size_t)row * H_ + h0);
        *(f4*)&Wcs[row][h0] = *(const f4*)(Wc + (size_t)row * H_ + h0);
    }
    for (int idx = tid; idx < B_ * P2_HS; idx += 1024)
        ((float*)hs)[idx] = 0.0f;
    __syncthreads();

    for (int t = 0; t < T_; ++t) {
        v8f ar[2], au[2];
#pragma unroll
        for (int j = 0; j < 2; ++j) {
            int col = nq * 32 + j * 16 + l16;
#pragma unroll
            for (int r = 0; r < 8; ++r) {
                int b = mt * 16 + r + lhi * 8;
                size_t off = ((size_t)b * T_ + t) * H_ + col;
                ar[j][r] = out[off];   // xr_t (includes bias)
                au[j][r] = xu[off];
            }
        }
#pragma unroll 4
        for (int ks = 0; ks < 32; ++ks) {
            v2f a;
            a.x = hs[mt * 16 + l16][ks * 4 + lhi * 2 + 0];
            a.y = hs[mt * 16 + l16][ks * 4 + lhi * 2 + 1];
            const int rk = ks * 4 + lhi * 2;
#pragma unroll
            for (int j = 0; j < 2; ++j) {
                int colb = nq * 32 + j * 16 + l16;
                v2f brf, buf;
                brf.x = Wrs[rk][colb]; brf.y = Wrs[rk + 1][colb];
                buf.x = Wus[rk][colb]; buf.y = Wus[rk + 1][colb];
                ar[j] = __builtin_amdgcn_wmma_f32_16x16x4_f32(
                    false, a, false, brf, (short)0, ar[j], false, false);
                au[j] = __builtin_amdgcn_wmma_f32_16x16x4_f32(
                    false, a, false, buf, (short)0, au[j], false, false);
            }
        }
        v8f hold[2], ug[2];
#pragma unroll
        for (int j = 0; j < 2; ++j) {
            int col = nq * 32 + j * 16 + l16;
#pragma unroll
            for (int r = 0; r < 8; ++r)
                hold[j][r] = hs[mt * 16 + r + lhi * 8][col];
        }
        __syncthreads();   // all hs reads done

#pragma unroll
        for (int j = 0; j < 2; ++j) {
            int col = nq * 32 + j * 16 + l16;
#pragma unroll
            for (int r = 0; r < 8; ++r) {
                int b = mt * 16 + r + lhi * 8;
                float rg = 1.0f / (1.0f + __expf(-ar[j][r]));
                ug[j][r] = 1.0f / (1.0f + __expf(-au[j][r]));
                hs[b][col] = rg * hold[j][r];   // hs <- r * h_old (in place)
            }
        }
        __syncthreads();   // rh ready

        v8f ac[2];
#pragma unroll
        for (int j = 0; j < 2; ++j) {
            int col = nq * 32 + j * 16 + l16;
#pragma unroll
            for (int r = 0; r < 8; ++r) {
                int b = mt * 16 + r + lhi * 8;
                ac[j][r] = xc[((size_t)b * T_ + t) * H_ + col];
            }
        }
#pragma unroll 4
        for (int ks = 0; ks < 32; ++ks) {
            v2f a;
            a.x = hs[mt * 16 + l16][ks * 4 + lhi * 2 + 0];
            a.y = hs[mt * 16 + l16][ks * 4 + lhi * 2 + 1];
            const int rk = ks * 4 + lhi * 2;
#pragma unroll
            for (int j = 0; j < 2; ++j) {
                int colb = nq * 32 + j * 16 + l16;
                v2f bcf;
                bcf.x = Wcs[rk][colb]; bcf.y = Wcs[rk + 1][colb];
                ac[j] = __builtin_amdgcn_wmma_f32_16x16x4_f32(
                    false, a, false, bcf, (short)0, ac[j], false, false);
            }
        }
        __syncthreads();   // rh reads done before overwrite

#pragma unroll
        for (int j = 0; j < 2; ++j) {
            int col = nq * 32 + j * 16 + l16;
#pragma unroll
            for (int r = 0; r < 8; ++r) {
                int b = mt * 16 + r + lhi * 8;
                float cg = tanhf(ac[j][r]);
                float u  = ug[j][r];
                float hn = hold[j][r] + u * (cg - hold[j][r]);
                hs[b][col] = hn;
                out[((size_t)b * T_ + t) * H_ + col] = hn;
            }
        }
        __syncthreads();   // h_new visible for next step
    }
}

extern "C" void kernel_launch(void* const* d_in, const int* in_sizes, int n_in,
                              void* d_out, int out_size, void* d_ws, size_t ws_size,
                              hipStream_t stream)
{
    const float* x  = (const float*)d_in[0];
    const float* Wr = (const float*)d_in[1];
    const float* br = (const float*)d_in[2];
    const float* Wu = (const float*)d_in[3];
    const float* bu = (const float*)d_in[4];
    const float* Wc = (const float*)d_in[5];
    const float* bc = (const float*)d_in[6];
    float* out = (float*)d_out;

    const size_t BTH = (size_t)B_ * T_ * H_;
    float*    xu  = (float*)d_ws;
    float*    xc  = xu + BTH;
    unsigned* WhP = (unsigned*)(xc + BTH);
    unsigned* WlP = WhP + (size_t)3 * KP_ * H_;

    const int prepTotal = 3 * KP_ * H_;
    gru_prep_w<<<(prepTotal + 255) / 256, 256, 0, stream>>>(Wr, Wu, Wc, WhP, WlP);
    gru_phase1<<<M_ / P1_MT, 256, 0, stream>>>(x, br, bu, bc, WhP, WlP,
                                               out /*xr*/, xu, xc);
    gru_phase2<<<1, 1024, 0, stream>>>(Wr, Wu, Wc, xu, xc, out);
}
